// TransformerBlock_16149077033199
// MI455X (gfx1250) — matrix operand order repaired and checked by reading, not compiled, not run on hardware
//
#include <hip/hip_runtime.h>
#include <math.h>

#define B_    4
#define N_    4096
#define C_    256
#define H_    512
#define NUM_  4
#define GS1_  128
#define KNN_  10
#define MT_   4   // M-tiles (16 rows each) per wave in the WMMA GEMM

typedef _Float16 v16h __attribute__((ext_vector_type(16)));
typedef _Float16 h8   __attribute__((ext_vector_type(8)));
typedef float    v8f  __attribute__((ext_vector_type(8)));

// ---------------------------------------------------------------------------
// Packing kernels
// ---------------------------------------------------------------------------

// featT_h[b][n][c] = (f16) features[b][c][n]
__global__ void pack_featT_kernel(const float* __restrict__ f, _Float16* __restrict__ o) {
  long long total = (long long)B_ * N_ * C_;
  for (long long i = blockIdx.x * (long long)blockDim.x + threadIdx.x; i < total;
       i += (long long)gridDim.x * blockDim.x) {
    long long b = i / ((long long)N_ * C_);
    long long r = i - b * (long long)N_ * C_;
    int n = (int)(r / C_), c = (int)(r % C_);
    o[i] = (_Float16)f[b * (long long)C_ * N_ + (long long)c * N_ + n];
  }
}

// dst[col*rows + row] = (f16)(w[row*cols + col] - (w2 ? w2[row*cols+col] : 0))
__global__ void pack_wt_kernel(const float* __restrict__ w, const float* __restrict__ w2,
                               _Float16* __restrict__ o, int rows, int cols) {
  long long total = (long long)rows * cols;
  for (long long i = blockIdx.x * (long long)blockDim.x + threadIdx.x; i < total;
       i += (long long)gridDim.x * blockDim.x) {
    int c = (int)(i / rows), r = (int)(i % rows);
    float v = w[(long long)r * cols + c];
    if (w2) v -= w2[(long long)r * cols + c];
    o[i] = (_Float16)v;
  }
}

// ---------------------------------------------------------------------------
// Batched WMMA GEMM: C[M,Nc] = A[M,K] * Bt[Nc,K]^T (+bias)(+acc)(relu)
// One wave computes a 64x16 macro-tile (MT_=4 stacked 16x16 tiles). The B
// fragment is loaded once per k-step and fed to 4 back-to-back WMMAs with the
// reuse_b operand-cache hint on all but the last. The k-loop is software-
// pipelined: fragments for step k+32 are prefetched before the WMMAs of step
// k issue, so the pre-WMMA s_wait_loadcnt only covers the older loads.
// Block = 8 waves = 64 rows x 128 cols. Rows addressed per-batch (row m ->
// batch m/rpb); rpb is a multiple of 64 so a macro-tile never spans batches.
// ---------------------------------------------------------------------------
__global__ __launch_bounds__(256)
void gemm_wmma_kernel(const _Float16* __restrict__ A, int lda, long long batchA,
                      const _Float16* __restrict__ Bt, int ldb,
                      float* __restrict__ C, int ldc, long long batchC,
                      _Float16* __restrict__ Ch,
                      const float* __restrict__ bias,
                      int rpb, int K, int relu_flag, int acc_flag) {
  const int lane = threadIdx.x & 31;
  const int wave = threadIdx.x >> 5;
  const int m0 = blockIdx.x * (16 * MT_);
  const int n0 = (blockIdx.y * 8 + wave) * 16;

  const int lrow = lane & 15;
  const int hi   = lane >> 4;

  const int b   = m0 / rpb;               // rpb % 64 == 0 -> whole macro-tile same batch
  const int ja0 = (m0 + lrow) - b * rpb;  // A row (in batch) for t = 0
  const _Float16* abase = A + (long long)b * batchA + (long long)ja0 * lda;
  const _Float16* bptr  = Bt + (long long)(n0 + lrow) * ldb;
  const long long astep = (long long)16 * lda;

  const int akb = hi * 16;   // A k-base within 32-chunk (contiguous k = 16h + s, matches B)
  const int bkb = hi * 16;   // B k-base within 32-chunk (contiguous k = 16h + s, matches A)

  v8f c[MT_];
  if (acc_flag) {
#pragma unroll
    for (int t = 0; t < MT_; ++t)
#pragma unroll
      for (int i = 0; i < 8; ++i) {
        int jc = (m0 + t * 16 + hi * 8 + i) - b * rpb;
        c[t][i] = C[(long long)b * batchC + (long long)jc * ldc + (n0 + lrow)];
      }
  } else {
#pragma unroll
    for (int t = 0; t < MT_; ++t)
#pragma unroll
      for (int i = 0; i < 8; ++i) c[t][i] = 0.0f;
  }

  // ---- prologue: load fragments for k = 0 ----
  v16h bf = *(const v16h*)(bptr + bkb);
  v16h a[MT_];
#pragma unroll
  for (int t = 0; t < MT_; ++t) {
    const _Float16* ap = abase + t * astep + akb;
    h8 a_lo = *(const h8*)(ap);
    h8 a_hi = *(const h8*)(ap + 8);
#pragma unroll
    for (int e = 0; e < 8; ++e) { a[t][e] = a_lo[e]; a[t][e + 8] = a_hi[e]; }
  }

  // ---- pipelined main loop: prefetch k+32 while computing k ----
#pragma unroll 2
  for (int k = 0; k + 32 < K; k += 32) {
    v16h bf_n = *(const v16h*)(bptr + (k + 32) + bkb);
    v16h an[MT_];
#pragma unroll
    for (int t = 0; t < MT_; ++t) {
      const _Float16* ap = abase + t * astep + (k + 32) + akb;
      h8 a_lo = *(const h8*)(ap);
      h8 a_hi = *(const h8*)(ap + 8);
#pragma unroll
      for (int e = 0; e < 8; ++e) { an[t][e] = a_lo[e]; an[t][e + 8] = a_hi[e]; }
    }
    c[0] = __builtin_amdgcn_wmma_f32_16x16x32_f16(false, a[0], false, bf, (short)0, c[0], false, true);
    c[1] = __builtin_amdgcn_wmma_f32_16x16x32_f16(false, a[1], false, bf, (short)0, c[1], false, true);
    c[2] = __builtin_amdgcn_wmma_f32_16x16x32_f16(false, a[2], false, bf, (short)0, c[2], false, true);
    c[3] = __builtin_amdgcn_wmma_f32_16x16x32_f16(false, a[3], false, bf, (short)0, c[3], false, false);
    bf = bf_n;
#pragma unroll
    for (int t = 0; t < MT_; ++t) a[t] = an[t];
  }

  // ---- epilogue: last k-step ----
  c[0] = __builtin_amdgcn_wmma_f32_16x16x32_f16(false, a[0], false, bf, (short)0, c[0], false, true);
  c[1] = __builtin_amdgcn_wmma_f32_16x16x32_f16(false, a[1], false, bf, (short)0, c[1], false, true);
  c[2] = __builtin_amdgcn_wmma_f32_16x16x32_f16(false, a[2], false, bf, (short)0, c[2], false, true);
  c[3] = __builtin_amdgcn_wmma_f32_16x16x32_f16(false, a[3], false, bf, (short)0, c[3], false, false);

  const int col = n0 + lrow;
  const float bv = bias ? bias[col] : 0.0f;
#pragma unroll
  for (int t = 0; t < MT_; ++t)
#pragma unroll
    for (int i = 0; i < 8; ++i) {
      int jc = (m0 + t * 16 + hi * 8 + i) - b * rpb;
      long long off = (long long)b * batchC + (long long)jc * ldc + col;
      float val = c[t][i] + bv;
      if (relu_flag) val = val > 0.0f ? val : 0.0f;
      if (C)  C[off]  = val;
      if (Ch) Ch[off] = (_Float16)val;
    }
}

// ---------------------------------------------------------------------------
// Column softmax over a strided axis, column cached in LDS.
// block = one (batch, col); data[b*bstride + i*stride + col], i in [0,len)
// ---------------------------------------------------------------------------
__global__ __launch_bounds__(256)
void softmax_col_kernel(float* __restrict__ data, int stride, int len,
                        long long bstride, int ncols, float scale) {
  __shared__ float cache[4096];
  __shared__ float red[256];
  const int b   = blockIdx.x / ncols;
  const int col = blockIdx.x % ncols;
  float* base = data + (long long)b * bstride + col;
  const int tid = threadIdx.x;

  float m = -3.0e38f;
  for (int i = tid; i < len; i += 256) {
    float v = base[(long long)i * stride];
    cache[i] = v;
    m = fmaxf(m, v);
  }
  red[tid] = m; __syncthreads();
  for (int s = 128; s > 0; s >>= 1) {
    if (tid < s) red[tid] = fmaxf(red[tid], red[tid + s]);
    __syncthreads();
  }
  const float mx = red[0];
  __syncthreads();

  float sum = 0.0f;
  for (int i = tid; i < len; i += 256) {
    float e = __expf((cache[i] - mx) * scale);
    cache[i] = e;
    sum += e;
  }
  red[tid] = sum; __syncthreads();
  for (int s = 128; s > 0; s >>= 1) {
    if (tid < s) red[tid] += red[tid + s];
    __syncthreads();
  }
  const float inv = 1.0f / red[0];
  __syncthreads();

  for (int i = tid; i < len; i += 256)
    base[(long long)i * stride] = cache[i] * inv;
}

// ---------------------------------------------------------------------------
// Brute-force KNN (top-10 smallest squared distance, ascending, stable ties).
// One thread = one query point; the batch's 4096 points staged in LDS (48 KB).
// ---------------------------------------------------------------------------
__global__ __launch_bounds__(256)
void knn_kernel(const float* __restrict__ xyz, int* __restrict__ out) {
  __shared__ float sx[N_], sy[N_], sz[N_];
  const int b = blockIdx.x >> 4;                  // N_/256 == 16 blocks per batch
  const int q = ((blockIdx.x & 15) << 8) + threadIdx.x;
  const float* base = xyz + (long long)b * N_ * 3;
  for (int i = threadIdx.x; i < N_; i += 256) {
    sx[i] = base[i * 3 + 0];
    sy[i] = base[i * 3 + 1];
    sz[i] = base[i * 3 + 2];
  }
  __syncthreads();
  const float qx = sx[q], qy = sy[q], qz = sz[q];
  float bd[KNN_]; int bi[KNN_];
#pragma unroll
  for (int k = 0; k < KNN_; ++k) { bd[k] = 3.0e38f; bi[k] = 0; }
  for (int m = 0; m < N_; ++m) {
    float dx = qx - sx[m], dy = qy - sy[m], dz = qz - sz[m];
    float d = dx * dx + dy * dy + dz * dz;
    if (d < bd[KNN_ - 1]) {
      float cd = d; int ci = m;
#pragma unroll
      for (int j = 0; j < KNN_; ++j) {
        if (cd < bd[j]) {
          float td = bd[j]; int ti = bi[j];
          bd[j] = cd; bi[j] = ci;
          cd = td; ci = ti;
        }
      }
    }
  }
  int* o = out + ((long long)b * N_ + q) * KNN_;
#pragma unroll
  for (int k = 0; k < KNN_; ++k) o[k] = bi[k];
}

// ---------------------------------------------------------------------------
// hmean_h[b,n,h] = (1/K) * sum_k relu( (xyz[n]-xyz[knn_k]) . fd1[:,h] + fd1_b[h] )
// One block per (b,n); neighbor diffs staged in LDS.
// ---------------------------------------------------------------------------
__global__ __launch_bounds__(256)
void posenc_kernel(const float* __restrict__ xyz, const int* __restrict__ knn,
                   const float* __restrict__ fd1, const float* __restrict__ fd1b,
                   _Float16* __restrict__ hmean_h) {
  const int b = blockIdx.x / N_;
  const int n = blockIdx.x % N_;
  __shared__ float dx[KNN_], dy[KNN_], dz[KNN_];
  const float* xb = xyz + (long long)b * N_ * 3;
  if (threadIdx.x < KNN_) {
    int idx = knn[((long long)b * N_ + n) * KNN_ + threadIdx.x];
    dx[threadIdx.x] = xb[n * 3 + 0] - xb[idx * 3 + 0];
    dy[threadIdx.x] = xb[n * 3 + 1] - xb[idx * 3 + 1];
    dz[threadIdx.x] = xb[n * 3 + 2] - xb[idx * 3 + 2];
  }
  __syncthreads();
  for (int h = threadIdx.x; h < H_; h += 256) {
    float w0 = fd1[h], w1 = fd1[H_ + h], w2 = fd1[2 * H_ + h], bb = fd1b[h];
    float acc = 0.0f;
#pragma unroll
    for (int k = 0; k < KNN_; ++k) {
      float t = dx[k] * w0 + dy[k] * w1 + dz[k] * w2 + bb;
      acc += t > 0.0f ? t : 0.0f;
    }
    hmean_h[((long long)b * N_ + n) * H_ + h] = (_Float16)(acc * (1.0f / KNN_));
  }
}

// res_h = (attn1 + tile(attn2)) * (v + pos_enc)   [vpos already holds v+pos]
__global__ void res_kernel(const float* __restrict__ attn1, const float* __restrict__ a2,
                           const float* __restrict__ vpos, _Float16* __restrict__ res_h) {
  long long total = (long long)B_ * N_ * H_;
  for (long long i = blockIdx.x * (long long)blockDim.x + threadIdx.x; i < total;
       i += (long long)gridDim.x * blockDim.x) {
    long long b = i / ((long long)N_ * H_);
    long long r = i - b * (long long)N_ * H_;
    int n = (int)(r / H_), h = (int)(r % H_);
    float attn = attn1[i] + a2[b * (long long)(N_ / NUM_) * H_ + (long long)(n & (N_ / NUM_ - 1)) * H_ + h];
    res_h[i] = (_Float16)(attn * vpos[i]);
  }
}

// out[b,c,n] = y[b,n,c] + features[b,c,n]; also append xyz to the output tuple
__global__ void final_kernel(const float* __restrict__ y, const float* __restrict__ feat,
                             const float* __restrict__ xyz, float* __restrict__ out) {
  long long total = (long long)B_ * C_ * N_;
  for (long long i = blockIdx.x * (long long)blockDim.x + threadIdx.x; i < total;
       i += (long long)gridDim.x * blockDim.x) {
    long long b = i / ((long long)C_ * N_);
    long long r = i - b * (long long)C_ * N_;
    int c = (int)(r / N_), n = (int)(r % N_);
    out[i] = y[(b * N_ + n) * (long long)C_ + c] + feat[i];
  }
  long long total2 = (long long)B_ * N_ * 3;
  for (long long i = blockIdx.x * (long long)blockDim.x + threadIdx.x; i < total2;
       i += (long long)gridDim.x * blockDim.x) {
    out[total + i] = xyz[i];
  }
}

// ---------------------------------------------------------------------------

extern "C" void kernel_launch(void* const* d_in, const int* in_sizes, int n_in,
                              void* d_out, int out_size, void* d_ws, size_t ws_size,
                              hipStream_t stream) {
  (void)in_sizes; (void)n_in; (void)out_size; (void)ws_size;

  const float* features = (const float*)d_in[0];
  const float* xyz    = (const float*)d_in[1];
  const float* fc1_w  = (const float*)d_in[2];
  const float* fc1_b  = (const float*)d_in[3];
  const float* fc2_w  = (const float*)d_in[4];
  const float* fc2_b  = (const float*)d_in[5];
  const float* fd1_w  = (const float*)d_in[6];
  const float* fd1_b  = (const float*)d_in[7];
  const float* fd2_w  = (const float*)d_in[8];
  const float* fd2_b  = (const float*)d_in[9];
  const float* fg1_w  = (const float*)d_in[10];
  const float* fg1_b  = (const float*)d_in[11];
  const float* fg2_w  = (const float*)d_in[12];
  const float* fg2_b  = (const float*)d_in[13];
  const float* fgh1_w = (const float*)d_in[14];
  const float* fgh1_b = (const float*)d_in[15];
  const float* fgh2_w = (const float*)d_in[16];
  const float* fgh2_b = (const float*)d_in[17];
  const float* wqs    = (const float*)d_in[18];
  const float* wks    = (const float*)d_in[19];
  const float* wvs    = (const float*)d_in[20];
  const float* wqs2   = (const float*)d_in[21];
  const float* wks2   = (const float*)d_in[22];

  size_t off = 0;
  auto alloc = [&](size_t bytes) -> void* {
    void* p = (char*)d_ws + off;
    off += (bytes + 255) & ~(size_t)255;
    return p;
  };

  const long long BNH = (long long)B_ * N_ * H_;            // 8,388,608
  const long long BQH = (long long)B_ * (N_ / NUM_) * H_;   // 2,097,152

  _Float16* featT   = (_Float16*)alloc((size_t)B_ * N_ * C_ * 2);
  _Float16* x_h     = (_Float16*)alloc((size_t)BNH * 2);
  _Float16* qk1_h   = (_Float16*)alloc((size_t)BNH * 2);
  _Float16* t1_h    = (_Float16*)alloc((size_t)BNH * 2);
  float*    a1      = (float*)   alloc((size_t)BNH * 4);
  _Float16* qk2_h   = (_Float16*)alloc((size_t)BQH * 2);
  _Float16* t2_h    = (_Float16*)alloc((size_t)BQH * 2);
  float*    a2      = (float*)   alloc((size_t)BQH * 4);
  float*    vbuf    = (float*)   alloc((size_t)BNH * 4);
  _Float16* hmean_h = (_Float16*)alloc((size_t)BNH * 2);
  _Float16* res_h   = (_Float16*)alloc((size_t)BNH * 2);
  float*    ybuf    = (float*)   alloc((size_t)B_ * N_ * C_ * 4);
  int*      knn_idx = (int*)     alloc((size_t)B_ * N_ * KNN_ * 4);
  _Float16* fc1_wt  = (_Float16*)alloc((size_t)C_ * H_ * 2);
  _Float16* wqk2_t  = (_Float16*)alloc((size_t)GS1_ * GS1_ * 2);
  _Float16* fgh1_t  = (_Float16*)alloc((size_t)GS1_ * GS1_ * 2);
  _Float16* fgh2_t  = (_Float16*)alloc((size_t)GS1_ * GS1_ * 2);
  _Float16* wqk_t   = (_Float16*)alloc((size_t)H_ * H_ * 2);
  _Float16* fg1_t   = (_Float16*)alloc((size_t)H_ * H_ * 2);
  _Float16* fg2_t   = (_Float16*)alloc((size_t)H_ * H_ * 2);
  _Float16* wvs_t   = (_Float16*)alloc((size_t)H_ * H_ * 2);
  _Float16* fd2_t   = (_Float16*)alloc((size_t)H_ * H_ * 2);
  _Float16* fc2_t   = (_Float16*)alloc((size_t)H_ * C_ * 2);

  // ---- Pack inputs / weights to f16 (B matrices transposed; q-k folded) ----
  pack_featT_kernel<<<4096, 256, 0, stream>>>(features, featT);
  pack_wt_kernel<<<512, 256, 0, stream>>>(fc1_w, nullptr, fc1_wt, C_, H_);
  pack_wt_kernel<<<64,  256, 0, stream>>>(wqs2, wks2, wqk2_t, GS1_, GS1_);
  pack_wt_kernel<<<64,  256, 0, stream>>>(fgh1_w, nullptr, fgh1_t, GS1_, GS1_);
  pack_wt_kernel<<<64,  256, 0, stream>>>(fgh2_w, nullptr, fgh2_t, GS1_, GS1_);
  pack_wt_kernel<<<1024, 256, 0, stream>>>(wqs, wks, wqk_t, H_, H_);
  pack_wt_kernel<<<1024, 256, 0, stream>>>(fg1_w, nullptr, fg1_t, H_, H_);
  pack_wt_kernel<<<1024, 256, 0, stream>>>(fg2_w, nullptr, fg2_t, H_, H_);
  pack_wt_kernel<<<1024, 256, 0, stream>>>(wvs, nullptr, wvs_t, H_, H_);
  pack_wt_kernel<<<1024, 256, 0, stream>>>(fd2_w, nullptr, fd2_t, H_, H_);
  pack_wt_kernel<<<512, 256, 0, stream>>>(fc2_w, nullptr, fc2_t, H_, C_);

  auto gemm = [&](const _Float16* A, int lda, long long batchA,
                  const _Float16* Bt, int ldb,
                  float* Cp, int ldc, long long batchC, _Float16* Ch,
                  const float* bias, int M, int Nc, int K, int rpb,
                  int relu, int acc) {
    dim3 g(M / (16 * MT_), Nc / 128);
    gemm_wmma_kernel<<<g, 256, 0, stream>>>(A, lda, batchA, Bt, ldb, Cp, ldc, batchC,
                                            Ch, bias, rpb, K, relu, acc);
  };

  // x = features^T @ fc1_w + fc1_b               (f16 output only)
  gemm(featT, C_, (long long)N_ * C_, fc1_wt, C_,
       nullptr, H_, (long long)N_ * H_, x_h, fc1_b,
       B_ * N_, H_, C_, N_, 0, 0);

  // Group attention path: (q1-k1) = xg @ (wqs2-wks2)   [M = B*N*NUM, 128x128]
  gemm(x_h, GS1_, (long long)N_ * H_, wqk2_t, GS1_,
       nullptr, GS1_, (long long)N_ * H_, qk1_h, nullptr,
       B_ * N_ * NUM_, GS1_, GS1_, N_ * NUM_, 0, 0);
  gemm(qk1_h, GS1_, (long long)N_ * H_, fgh1_t, GS1_,
       nullptr, GS1_, (long long)N_ * H_, t1_h, fgh1_b,
       B_ * N_ * NUM_, GS1_, GS1_, N_ * NUM_, 1, 0);
  gemm(t1_h, GS1_, (long long)N_ * H_, fgh2_t, GS1_,
       a1, GS1_, (long long)N_ * H_, nullptr, fgh2_b,
       B_ * N_ * NUM_, GS1_, GS1_, N_ * NUM_, 0, 0);
  // softmax over N (stride H) with 1/sqrt(GS1)
  softmax_col_kernel<<<B_ * H_, 256, 0, stream>>>(a1, H_, N_, (long long)N_ * H_, H_,
                                                  1.0f / sqrtf((float)GS1_));

  // Global attention path on last quarter: (q2-k2) = gf2 @ (wqs-wks)
  gemm(x_h + (size_t)(NUM_ - 1) * (N_ / NUM_) * H_, H_, (long long)N_ * H_, wqk_t, H_,
       nullptr, H_, (long long)(N_ / NUM_) * H_, qk2_h, nullptr,
       B_ * (N_ / NUM_), H_, H_, N_ / NUM_, 0, 0);
  gemm(qk2_h, H_, (long long)(N_ / NUM_) * H_, fg1_t, H_,
       nullptr, H_, (long long)(N_ / NUM_) * H_, t2_h, fg1_b,
       B_ * (N_ / NUM_), H_, H_, N_ / NUM_, 1, 0);
  gemm(t2_h, H_, (long long)(N_ / NUM_) * H_, fg2_t, H_,
       a2, H_, (long long)(N_ / NUM_) * H_, nullptr, fg2_b,
       B_ * (N_ / NUM_), H_, H_, N_ / NUM_, 0, 0);
  // softmax over the 1024-token axis (stride H) with 1/sqrt(H)
  softmax_col_kernel<<<B_ * H_, 256, 0, stream>>>(a2, H_, N_ / NUM_,
                                                  (long long)(N_ / NUM_) * H_, H_,
                                                  1.0f / sqrtf((float)H_));

  // v = x @ wvs
  gemm(x_h, H_, (long long)N_ * H_, wvs_t, H_,
       vbuf, H_, (long long)N_ * H_, nullptr, nullptr,
       B_ * N_, H_, H_, N_, 0, 0);

  // KNN + positional encoding (mean over K commutes with fd2)
  knn_kernel<<<B_ * (N_ / 256), 256, 0, stream>>>(xyz, knn_idx);
  posenc_kernel<<<B_ * N_, 256, 0, stream>>>(xyz, knn_idx, fd1_w, fd1_b, hmean_h);
  // v += hmean @ fd2 + fd2_b     (accumulate into v)
  gemm(hmean_h, H_, (long long)N_ * H_, fd2_t, H_,
       vbuf, H_, (long long)N_ * H_, nullptr, fd2_b,
       B_ * N_, H_, H_, N_, 0, 1);

  // res = (attn1 + tile(attn2)) * (v + pos_enc)
  res_kernel<<<4096, 256, 0, stream>>>(a1, a2, vbuf, res_h);

  // y = res @ fc2_w + fc2_b
  gemm(res_h, H_, (long long)N_ * H_, fc2_t, H_,
       ybuf, C_, (long long)N_ * C_, nullptr, fc2_b,
       B_ * N_, C_, H_, N_, 0, 0);

  // out = y^T + features ; append xyz
  final_kernel<<<4096, 256, 0, stream>>>(ybuf, features, xyz, (float*)d_out);
}
